// TextUnit_20323785244717
// MI455X (gfx1250) — matrix operand order repaired and checked by reading, not compiled, not run on hardware
//
#include <hip/hip_runtime.h>

#define D     768
#define NENT  32
#define BMEN  16
#define LTOK  128
#define SLEN  129
#define NEXP  8

typedef __attribute__((ext_vector_type(16))) __bf16 v16bf;
typedef __attribute__((ext_vector_type(8)))  float  v8f;
typedef unsigned short u16;

// ---------------- helpers ----------------

__device__ __forceinline__ u16 f2bf(float f) {
  unsigned u = __float_as_uint(f);
  unsigned r = (u + 0x7FFFu + ((u >> 16) & 1u)) >> 16;   // RNE
  return (u16)r;
}

// A operand (16x32 bf16): lane holds row = lane%16.
// halves 0..7  = K[kBase + g*8 .. +7], halves 8..15 = K[kBase+16+g*8 .. +7]
__device__ __forceinline__ v16bf load_a16(const u16* __restrict__ rowPtr, int kBase, int g) {
  union { v16bf v; uint4 q[2]; } u;
  u.q[0] = *(const uint4*)(rowPtr + kBase + g * 8);
  u.q[1] = *(const uint4*)(rowPtr + kBase + 16 + g * 8);
  return u.v;
}

// B operand (32x16 bf16): lane holds col = lane%16; halves 0..7 = K[kBase+g*8 .. +7], halves 8..15 = K[kBase+16+g*8 .. +7]
__device__ __forceinline__ v16bf load_b16(const u16* __restrict__ rowPtr, int kBase, int g) {
  union { v16bf v; uint4 q[2]; } u;
  u.q[0] = *(const uint4*)(rowPtr + kBase + g * 8);
  u.q[1] = *(const uint4*)(rowPtr + kBase + 16 + g * 8);
  return u.v;
}

// ---------------- conversion kernels ----------------

// in: (nMat, D, D) fp32 stored (d,o) -> out: (nMat, D, D) bf16 stored (o,d)
__global__ void k_wT(const float* __restrict__ in, u16* __restrict__ out, int nMat) {
  long idx = (long)blockIdx.x * blockDim.x + threadIdx.x;
  long tot = (long)nMat * D * D;
  if (idx >= tot) return;
  int m = (int)(idx / ((long)D * D));
  int r = (int)(idx % ((long)D * D));
  int o = r / D, d = r % D;
  out[(long)m * D * D + (long)o * D + d] = f2bf(in[(long)m * D * D + (long)d * D + o]);
}

// concat([cls, tokens]) rows -> bf16 [nItems*SLEN][D]
__global__ void k_xcat(const float* __restrict__ cls, const float* __restrict__ tok,
                       u16* __restrict__ out, int nItems) {
  long idx = (long)blockIdx.x * blockDim.x + threadIdx.x;
  long tot = (long)nItems * SLEN * D;
  if (idx >= tot) return;
  int t = (int)(idx / D), j = (int)(idx % D);
  int n = t / SLEN, s = t % SLEN;
  float v = (s == 0) ? cls[(long)n * D + j]
                     : tok[((long)n * LTOK + (s - 1)) * (long)D + j];
  out[(long)t * D + j] = f2bf(v);
}

// ---------------- gating (fp32 exact; discrete top-k must not flip) ----------------
__global__ void k_gate(const float* __restrict__ cls, const float* __restrict__ tok,
                       const float* __restrict__ gw, const float* __restrict__ gb,
                       float* __restrict__ wOut, int nItems) {
  int t = blockIdx.x * blockDim.x + threadIdx.x;
  if (t >= nItems * SLEN) return;
  int n = t / SLEN, s = t % SLEN;
  const float* x = (s == 0) ? (cls + (long)n * D) : (tok + ((long)n * LTOK + s - 1) * (long)D);
  float lg[NEXP];
#pragma unroll
  for (int e = 0; e < NEXP; e++) lg[e] = gb[e];
  for (int d = 0; d < D; d++) {
    float xv = x[d];
#pragma unroll
    for (int e = 0; e < NEXP; e++) lg[e] += xv * gw[d * NEXP + e];
  }
  float mx = lg[0];
#pragma unroll
  for (int e = 1; e < NEXP; e++) mx = fmaxf(mx, lg[e]);
  float pe[NEXP];
#pragma unroll
  for (int e = 0; e < NEXP; e++) pe[e] = __expf(lg[e] - mx);
  int i1 = 0;
#pragma unroll
  for (int e = 1; e < NEXP; e++) if (pe[e] > pe[i1]) i1 = e;      // earliest on tie
  int i2 = (i1 == 0) ? 1 : 0;
#pragma unroll
  for (int e = 0; e < NEXP; e++) if (e != i1 && pe[e] > pe[i2]) i2 = e;
  float denom = pe[i1] + pe[i2];
#pragma unroll
  for (int e = 0; e < NEXP; e++) {
    float w = (e == i1) ? pe[i1] / denom : ((e == i2) ? pe[i2] / denom : 0.0f);
    wOut[(long)t * NEXP + e] = w;
  }
}

// ---------------- MoE GEMM: Y[t,o] = sum_e w[t,e]*(X[t,:]@W_e[:,o] + b_e[o]) ----------------
// block (32,4): each wave one 16x16 output tile; grid (T/16, 12) -> 48 col tiles.
// Expert loop is INNERMOST: A loaded once per K-step, 8 live accumulators (no spills).
__global__ void k_moe(const u16* __restrict__ Xb, const u16* __restrict__ WT,
                      const float* __restrict__ gateW, const float* __restrict__ expB,
                      float* __restrict__ Yf, u16* __restrict__ Yb) {
  int lane = threadIdx.x;
  int g = lane >> 4, nc = lane & 15;
  int rowBase = blockIdx.x * 16;
  int colTile = blockIdx.y * blockDim.y + threadIdx.y;
  int o = colTile * 16 + nc;
  const u16* arow = Xb + (long)(rowBase + nc) * D;
  const u16* bbase = WT + (long)o * D;   // expert e at + e*D*D
  v8f zero = {};
  v8f P[NEXP];
#pragma unroll
  for (int e = 0; e < NEXP; e++) P[e] = zero;
#pragma unroll 2
  for (int k = 0; k < D; k += 32) {
    v16bf a = load_a16(arow, k, g);
#pragma unroll
    for (int e = 0; e < NEXP; e++) {
      v16bf b = load_b16(bbase + (long)e * D * D, k, g);
      P[e] = __builtin_amdgcn_wmma_f32_16x16x32_bf16(false, a, false, b, (short)0, P[e], false, false);
    }
  }
  float acc[8];
#pragma unroll
  for (int r = 0; r < 8; r++) acc[r] = 0.0f;
#pragma unroll
  for (int e = 0; e < NEXP; e++) {
    float be = expB[e * D + o];
#pragma unroll
    for (int r = 0; r < 8; r++) {
      float w = gateW[(long)(rowBase + g * 8 + r) * NEXP + e];
      acc[r] += w * (P[e][r] + be);
    }
  }
#pragma unroll
  for (int r = 0; r < 8; r++) {
    long row = rowBase + g * 8 + r;
    float v = acc[r];
    if (Yf) Yf[row * D + o] = v;
    if (Yb) Yb[row * D + o] = f2bf(v);
  }
}

// ---------------- linear GEMM: Y[t,o] = X[t,:]@W[:,o] + b[o] ----------------
__global__ void k_lin(const u16* __restrict__ Xb, const u16* __restrict__ WT,
                      const float* __restrict__ bias,
                      float* __restrict__ Yf, u16* __restrict__ Yb) {
  int lane = threadIdx.x;
  int g = lane >> 4, nc = lane & 15;
  int rowBase = blockIdx.x * 16;
  int colTile = blockIdx.y * blockDim.y + threadIdx.y;
  int o = colTile * 16 + nc;
  const u16* arow = Xb + (long)(rowBase + nc) * D;
  const u16* brow = WT + (long)o * D;
  v8f p = {};
  for (int k = 0; k < D; k += 32) {
    v16bf a = load_a16(arow, k, g);
    v16bf b = load_b16(brow, k, g);
    p = __builtin_amdgcn_wmma_f32_16x16x32_bf16(false, a, false, b, (short)0, p, false, false);
  }
  float bo = bias[o];
#pragma unroll
  for (int r = 0; r < 8; r++) {
    long row = rowBase + g * 8 + r;
    float v = p[r] + bo;
    if (Yf) Yf[row * D + o] = v;
    if (Yb) Yb[row * D + o] = f2bf(v);
  }
}

// ---------------- fused attention/softmax/mean/LN/dots per (b,n) pair ----------------
__device__ __forceinline__ float blk_sum(float v, float* sred) {
#pragma unroll
  for (int m = 16; m >= 1; m >>= 1) v += __shfl_xor(v, m, 32);
  int w = threadIdx.x >> 5;
  __syncthreads();                       // protect sred reuse across calls
  if ((threadIdx.x & 31) == 0) sred[w] = v;
  __syncthreads();
  float t = sred[0] + sred[1] + sred[2] + sred[3] + sred[4] + sred[5] + sred[6] + sred[7];
  return t;
}

__global__ __launch_bounds__(256)
void k_attn(const u16* __restrict__ qB, const u16* __restrict__ kB,
            const float* __restrict__ vF, const float* __restrict__ moeEntF,
            const float* __restrict__ eCls, const float* __restrict__ mCls,
            const float* __restrict__ lnw, const float* __restrict__ lnb,
            float* __restrict__ out) {
  __shared__ float cSum[LTOK];
  __shared__ float sred[8];
  int tid = threadIdx.x;
  int bI = blockIdx.x, nI = blockIdx.y;
  int wave = tid >> 5, lane = tid & 31, g = lane >> 4, nc = lane & 15;
  if (tid < LTOK) cSum[tid] = 0.0f;
  __syncthreads();

  // scores: wave owns rows l = 16*wave .. +15, all 128 cols (8 accum tiles)
  const u16* qrow = qB + (long)(nI * SLEN + 1 + wave * 16 + nc) * D;
  const u16* kbase = kB + (long)(bI * SLEN + 1) * D;
  v8f zero = {};
  v8f acc[8];
#pragma unroll
  for (int tc = 0; tc < 8; tc++) acc[tc] = zero;
#pragma unroll 2
  for (int k = 0; k < D; k += 32) {
    v16bf a = load_a16(qrow, k, g);
#pragma unroll
    for (int tc = 0; tc < 8; tc++) {
      v16bf b = load_b16(kbase + (long)(tc * 16 + nc) * D, k, g);
      acc[tc] = __builtin_amdgcn_wmma_f32_16x16x32_bf16(false, a, false, b, (short)0, acc[tc], false, false);
    }
  }

  // row softmax fully in registers (row = 16*wave + g*8 + r, cols across 8 tiles x 16 lanes)
  const float invs = 0.03608439182435161f;   // 1/sqrt(768)
#pragma unroll
  for (int r = 0; r < 8; r++) {
    float mx = -3.0e38f;
#pragma unroll
    for (int tc = 0; tc < 8; tc++) mx = fmaxf(mx, acc[tc][r]);
#pragma unroll
    for (int m = 8; m >= 1; m >>= 1) mx = fmaxf(mx, __shfl_xor(mx, m, 32));
    float s = 0.0f;
#pragma unroll
    for (int tc = 0; tc < 8; tc++) {
      float e = __expf((acc[tc][r] - mx) * invs);
      acc[tc][r] = e;
      s += e;
    }
#pragma unroll
    for (int m = 8; m >= 1; m >>= 1) s += __shfl_xor(s, m, 32);
    float rs = 1.0f / s;
#pragma unroll
    for (int tc = 0; tc < 8; tc++) acc[tc][r] *= rs;
  }

  // column sums of probs (mean over l collapses the context einsum)
#pragma unroll
  for (int tc = 0; tc < 8; tc++) {
    float sp = 0.0f;
#pragma unroll
    for (int r = 0; r < 8; r++) sp += acc[tc][r];
    sp += __shfl_xor(sp, 16, 32);
    if (lane < 16) atomicAdd(&cSum[tc * 16 + nc], sp);
  }
  __syncthreads();

  // ctx[d] = (1/128) * sum_m cSum[m] * v[b, m, d]   (fp32)
  const float* vrow = vF + (long)(bI * SLEN + 1) * D;
  float ctx[3];
#pragma unroll
  for (int j = 0; j < 3; j++) {
    int d = tid + j * 256;
    float s = 0.0f;
    for (int m = 0; m < LTOK; m++) s += cSum[m] * vrow[(long)m * D + d];
    ctx[j] = s * (1.0f / 128.0f);
  }

  // layernorm + g2l dot + g2g dot
  float mu = blk_sum(ctx[0] + ctx[1] + ctx[2], sred) * (1.0f / 768.0f);
  float vp = 0.0f;
#pragma unroll
  for (int j = 0; j < 3; j++) { float dd = ctx[j] - mu; vp += dd * dd; }
  float var = blk_sum(vp, sred) * (1.0f / 768.0f);
  float rstd = rsqrtf(var + 1e-5f);
  const float* cm = moeEntF + (long)(nI * SLEN) * D;   // ent_cls_moe row (s==0)
  float d1 = 0.0f, d2 = 0.0f;
#pragma unroll
  for (int j = 0; j < 3; j++) {
    int d = tid + j * 256;
    float y = (ctx[j] - mu) * rstd * lnw[d] + lnb[d];
    d1 += y * cm[d];
    d2 += mCls[(long)bI * D + d] * eCls[(long)nI * D + d];
  }
  float g2l = blk_sum(d1, sred);
  float g2g = blk_sum(d2, sred);
  if (tid == 0) out[bI * 32 + nI] = 0.5f * (g2l + g2g);
}

// ---------------- launcher ----------------

extern "C" void kernel_launch(void* const* d_in, const int* in_sizes, int n_in,
                              void* d_out, int out_size, void* d_ws, size_t ws_size,
                              hipStream_t stream) {
  (void)in_sizes; (void)n_in; (void)out_size; (void)ws_size;
  const float* eCls = (const float*)d_in[0];
  const float* eTok = (const float*)d_in[1];
  const float* mCls = (const float*)d_in[2];
  const float* mTok = (const float*)d_in[3];
  const float* gw   = (const float*)d_in[4];
  const float* gb   = (const float*)d_in[5];
  const float* expW = (const float*)d_in[6];
  const float* expB = (const float*)d_in[7];
  const float* wq   = (const float*)d_in[8];
  const float* bq   = (const float*)d_in[9];
  const float* wk   = (const float*)d_in[10];
  const float* bk   = (const float*)d_in[11];
  const float* wv   = (const float*)d_in[12];
  const float* bv   = (const float*)d_in[13];
  const float* lnw  = (const float*)d_in[14];
  const float* lnb  = (const float*)d_in[15];
  float* out = (float*)d_out;

  char* p = (char*)d_ws;
  auto carve = [&](size_t bytes) -> void* {
    void* r = (void*)p;
    p += (bytes + 255) & ~((size_t)255);
    return r;
  };
  const int TE = NENT * SLEN;  // 4128 (= 258*16)
  const int TM = BMEN * SLEN;  // 2064 (= 129*16)

  u16*   XeB   = (u16*)  carve((size_t)TE * D * 2);
  u16*   XmB   = (u16*)  carve((size_t)TM * D * 2);
  u16*   eWT   = (u16*)  carve((size_t)NEXP * D * D * 2);
  u16*   qWT   = (u16*)  carve((size_t)D * D * 2);
  u16*   kWT   = (u16*)  carve((size_t)D * D * 2);
  u16*   vWT   = (u16*)  carve((size_t)D * D * 2);
  float* gE    = (float*)carve((size_t)TE * NEXP * 4);
  float* gM    = (float*)carve((size_t)TM * NEXP * 4);
  float* moeEF = (float*)carve((size_t)TE * D * 4);
  u16*   moeEB = (u16*)  carve((size_t)TE * D * 2);
  u16*   moeMB = (u16*)  carve((size_t)TM * D * 2);
  u16*   qB    = (u16*)  carve((size_t)TE * D * 2);
  u16*   kB    = (u16*)  carve((size_t)TM * D * 2);
  float* vFp   = (float*)carve((size_t)TM * D * 4);

  // weight transpose + bf16 convert
  {
    long tot = (long)NEXP * D * D;
    k_wT<<<(unsigned)((tot + 255) / 256), 256, 0, stream>>>(expW, eWT, NEXP);
  }
  {
    long tot = (long)D * D;
    unsigned nb = (unsigned)((tot + 255) / 256);
    k_wT<<<nb, 256, 0, stream>>>(wq, qWT, 1);
    k_wT<<<nb, 256, 0, stream>>>(wk, kWT, 1);
    k_wT<<<nb, 256, 0, stream>>>(wv, vWT, 1);
  }
  // concat + bf16 convert of activations
  k_xcat<<<(unsigned)(((long)TE * D + 255) / 256), 256, 0, stream>>>(eCls, eTok, XeB, NENT);
  k_xcat<<<(unsigned)(((long)TM * D + 255) / 256), 256, 0, stream>>>(mCls, mTok, XmB, BMEN);
  // gating in fp32
  k_gate<<<(TE + 127) / 128, 128, 0, stream>>>(eCls, eTok, gw, gb, gE, NENT);
  k_gate<<<(TM + 127) / 128, 128, 0, stream>>>(mCls, mTok, gw, gb, gM, BMEN);
  // MoE GEMMs
  dim3 blk(32, 4);
  k_moe<<<dim3(TE / 16, 12), blk, 0, stream>>>(XeB, eWT, gE, expB, moeEF, moeEB);
  k_moe<<<dim3(TM / 16, 12), blk, 0, stream>>>(XmB, eWT, gM, expB, nullptr, moeMB);
  // Q/K/V GEMMs (computed over all 129 rows; attention indexes rows 1..128)
  k_lin<<<dim3(TE / 16, 12), blk, 0, stream>>>(moeEB, qWT, bq, nullptr, qB);
  k_lin<<<dim3(TM / 16, 12), blk, 0, stream>>>(moeMB, kWT, bk, nullptr, kB);
  k_lin<<<dim3(TM / 16, 12), blk, 0, stream>>>(moeMB, vWT, bv, vFp, nullptr);
  // fused attention + epilogue
  k_attn<<<dim3(BMEN, NENT), 256, 0, stream>>>(qB, kB, vFp, moeEF, eCls, mCls, lnw, lnb, out);
}
